// STGNN_29222957482126
// MI455X (gfx1250) — matrix operand order repaired and checked by reading, not compiled, not run on hardware
//
#include <hip/hip_runtime.h>

typedef __attribute__((ext_vector_type(16))) _Float16 v16h;
typedef __attribute__((ext_vector_type(8)))  float    v8f;

#define D_IN  32
#define D_H   64
#define D_OUT 16

// ---------------- init: zero accumulators, deg = 1 (self-loop) ----------------
__global__ void gcn_init(float* __restrict__ agg1, float* __restrict__ out,
                         float* __restrict__ deg, int N) {
  int idx = blockIdx.x * blockDim.x + threadIdx.x;
  if (idx < N * D_H)   agg1[idx] = 0.0f;
  if (idx < N * D_OUT) out[idx]  = 0.0f;
  if (idx < N)         deg[idx]  = 1.0f;   // self-loop contribution
}

// ---------------- degree: atomic histogram over dst ----------------
__global__ void gcn_deg(const long long* __restrict__ dst,
                        float* __restrict__ deg, int E) {
  int e = blockIdx.x * blockDim.x + threadIdx.x;
  if (e < E) unsafeAtomicAdd(&deg[(int)dst[e]], 1.0f);
}

__global__ void gcn_rsqrt(float* __restrict__ dinv, int N) {
  int i = blockIdx.x * blockDim.x + threadIdx.x;
  if (i < N) dinv[i] = rsqrtf(dinv[i]);
}

// ---------------- GEMM1: h1[N,64] = x[N,32] @ W1[32,64] via WMMA ----------------
// One wave handles a 16-node row tile; K=32 == one v_wmma_f32_16x16x32_f16.
__global__ void gcn_gemm1(const float* __restrict__ x, const float* __restrict__ W,
                          float* __restrict__ h, int N) {
  int wave = (int)((blockIdx.x * blockDim.x + threadIdx.x) >> 5);
  int lane = threadIdx.x & 31;
  if (wave * 16 >= N) return;                 // wave-uniform: EXEC stays all-1s
  int half = lane >> 4;                       // lane half selects K sub-block
  int m    = lane & 15;                       // A row / B,C column
  int row  = wave * 16 + m;

  // A: 16x32 f16, contiguous layout: K = half*16+2v (matches B)
  v16h a;
#pragma unroll
  for (int v = 0; v < 8; ++v) {
    int k0 = half * 16 + 2 * v;
    a[2 * v]     = (_Float16)x[row * D_IN + k0];
    a[2 * v + 1] = (_Float16)x[row * D_IN + k0 + 1];
  }

#pragma unroll
  for (int ct = 0; ct < 4; ++ct) {            // 4 column tiles cover D_H = 64
    // B: 32x16 f16; lanes 0-15 hold K=0..15 (2/VGPR), lanes 16-31 hold K=16..31
    v16h b;
#pragma unroll
    for (int v = 0; v < 8; ++v) {
      int k = half * 16 + 2 * v;
      b[2 * v]     = (_Float16)W[k * D_H + ct * 16 + m];
      b[2 * v + 1] = (_Float16)W[(k + 1) * D_H + ct * 16 + m];
    }
    v8f c = {};
    c = __builtin_amdgcn_wmma_f32_16x16x32_f16(false, a, false, b,
                                               (short)0, c, false, false);
    // C/D: VGPR v holds row (v + half*8), lane&15 is column
#pragma unroll
    for (int v = 0; v < 8; ++v)
      h[(wave * 16 + v + half * 8) * D_H + ct * 16 + m] = c[v];
  }
}

// ---------------- scatter layer 1: 32 lanes per edge, float2 per lane ----------------
__global__ void gcn_scatter1(const long long* __restrict__ src,
                             const long long* __restrict__ dst,
                             const float* __restrict__ dinv,
                             const float* __restrict__ h1,
                             float* __restrict__ agg1, int E) {
  long long t = (long long)blockIdx.x * blockDim.x + threadIdx.x;
  int e = (int)(t >> 5);
  int lane = (int)(t & 31);
  if (e >= E) return;
  int s = (int)src[e], d = (int)dst[e];
  float nrm = dinv[s] * dinv[d];
  const float2 v = *(const float2*)(h1 + s * D_H + lane * 2);
  unsafeAtomicAdd(agg1 + d * D_H + lane * 2,     v.x * nrm);
  unsafeAtomicAdd(agg1 + d * D_H + lane * 2 + 1, v.y * nrm);
}

// ---------------- self-loop + bias + ReLU (in place on agg1) ----------------
__global__ void gcn_relu1(const float* __restrict__ h1, const float* __restrict__ dinv,
                          const float* __restrict__ b1, float* __restrict__ agg1, int N) {
  int idx = blockIdx.x * blockDim.x + threadIdx.x;
  if (idx >= N * D_H) return;
  int i = idx >> 6, f = idx & 63;
  float di = dinv[i];
  float z = agg1[idx] + h1[idx] * di * di + b1[f];
  agg1[idx] = fmaxf(z, 0.0f);
}

// ---------------- GEMM2: h2[N,16] = z1[N,64] @ W2[64,16], two chained WMMAs ----------------
__global__ void gcn_gemm2(const float* __restrict__ z, const float* __restrict__ W,
                          float* __restrict__ h2, int N) {
  int wave = (int)((blockIdx.x * blockDim.x + threadIdx.x) >> 5);
  int lane = threadIdx.x & 31;
  if (wave * 16 >= N) return;
  int half = lane >> 4;
  int m    = lane & 15;
  int row  = wave * 16 + m;

  v16h a0, a1, b0, b1;
#pragma unroll
  for (int v = 0; v < 8; ++v) {
    int k0 = half * 16 + 2 * v;
    a0[2 * v]     = (_Float16)z[row * D_H + k0];
    a0[2 * v + 1] = (_Float16)z[row * D_H + k0 + 1];
    a1[2 * v]     = (_Float16)z[row * D_H + 32 + k0];
    a1[2 * v + 1] = (_Float16)z[row * D_H + 32 + k0 + 1];
  }
#pragma unroll
  for (int v = 0; v < 8; ++v) {
    int k = half * 16 + 2 * v;
    b0[2 * v]     = (_Float16)W[k * D_OUT + m];
    b0[2 * v + 1] = (_Float16)W[(k + 1) * D_OUT + m];
    b1[2 * v]     = (_Float16)W[(k + 32) * D_OUT + m];
    b1[2 * v + 1] = (_Float16)W[(k + 33) * D_OUT + m];
  }
  v8f c = {};
  c = __builtin_amdgcn_wmma_f32_16x16x32_f16(false, a0, false, b0, (short)0, c, false, false);
  c = __builtin_amdgcn_wmma_f32_16x16x32_f16(false, a1, false, b1, (short)0, c, false, false);
#pragma unroll
  for (int v = 0; v < 8; ++v)
    h2[(wave * 16 + v + half * 8) * D_OUT + m] = c[v];
}

// ---------------- scatter layer 2: 16 lanes per edge ----------------
__global__ void gcn_scatter2(const long long* __restrict__ src,
                             const long long* __restrict__ dst,
                             const float* __restrict__ dinv,
                             const float* __restrict__ h2,
                             float* __restrict__ out, int E) {
  long long t = (long long)blockIdx.x * blockDim.x + threadIdx.x;
  int e = (int)(t >> 4);
  int f = (int)(t & 15);
  if (e >= E) return;
  int s = (int)src[e], d = (int)dst[e];
  float nrm = dinv[s] * dinv[d];
  unsafeAtomicAdd(out + d * D_OUT + f, h2[s * D_OUT + f] * nrm);
}

// ---------------- final: self-loop + bias on output ----------------
__global__ void gcn_final(const float* __restrict__ h2, const float* __restrict__ dinv,
                          const float* __restrict__ b2, float* __restrict__ out, int N) {
  int idx = blockIdx.x * blockDim.x + threadIdx.x;
  if (idx >= N * D_OUT) return;
  int i = idx >> 4, f = idx & 15;
  float di = dinv[i];
  out[idx] += h2[idx] * di * di + b2[f];
}

extern "C" void kernel_launch(void* const* d_in, const int* in_sizes, int n_in,
                              void* d_out, int out_size, void* d_ws, size_t ws_size,
                              hipStream_t stream) {
  const float*     x   = (const float*)d_in[0];
  const long long* ei  = (const long long*)d_in[1];   // int64 [2, E]
  const float*     W1  = (const float*)d_in[2];
  const float*     b1  = (const float*)d_in[3];
  const float*     W2  = (const float*)d_in[4];
  const float*     b2  = (const float*)d_in[5];
  float* out = (float*)d_out;

  const int N = in_sizes[0] / D_IN;    // 100000
  const int E = in_sizes[1] / 2;       // 1600000
  const long long* src = ei;
  const long long* dst = ei + E;

  // workspace layout (floats): dinv[N] | h1[N*64] | agg1[N*64] | h2[N*16]
  float* ws   = (float*)d_ws;
  float* dinv = ws;
  float* h1   = ws + ((N + 255) & ~255);
  float* agg1 = h1 + (size_t)N * D_H;
  float* h2   = agg1 + (size_t)N * D_H;

  const int B = 256;
  int g_init  = (N * D_H + B - 1) / B;
  int g_deg   = (E + B - 1) / B;
  int g_node  = (N + B - 1) / B;
  int g_gemm  = (((N + 15) / 16) * 32 + B - 1) / B;   // one wave per 16-row tile
  long long t1 = (long long)E * 32;
  long long t2 = (long long)E * 16;
  int g_sc1   = (int)((t1 + B - 1) / B);
  int g_sc2   = (int)((t2 + B - 1) / B);
  int g_nh    = (N * D_H + B - 1) / B;
  int g_no    = (N * D_OUT + B - 1) / B;

  gcn_init    <<<g_init, B, 0, stream>>>(agg1, out, dinv, N);
  gcn_deg     <<<g_deg,  B, 0, stream>>>(dst, dinv, E);
  gcn_rsqrt   <<<g_node, B, 0, stream>>>(dinv, N);
  gcn_gemm1   <<<g_gemm, B, 0, stream>>>(x, W1, h1, N);
  gcn_scatter1<<<g_sc1,  B, 0, stream>>>(src, dst, dinv, h1, agg1, E);
  gcn_relu1   <<<g_nh,   B, 0, stream>>>(h1, dinv, b1, agg1, N);
  gcn_gemm2   <<<g_gemm, B, 0, stream>>>(agg1, W2, h2, N);
  gcn_scatter2<<<g_sc2,  B, 0, stream>>>(src, dst, dinv, h2, out, E);
  gcn_final   <<<g_no,   B, 0, stream>>>(h2, dinv, b2, out, N);
}
